// TransformerBlock_2130303779083
// MI455X (gfx1250) — matrix operand order repaired and checked by reading, not compiled, not run on hardware
//
#include <hip/hip_runtime.h>

// ---------------------------------------------------------------------------
// CDNA5 / gfx1250 transformer block.
//   - weights pre-converted fp32 -> f16 into d_ws in WMMA B-fragment layout
//   - main kernel: 1 workgroup (4 wave32) per batch element, all GEMMs via
//     v_wmma_f32_16x16x32_f16, activations staged in LDS.
// d_ws usage: 6144 fragments * 512 halves * 2B = 6,291,456 bytes.
// ---------------------------------------------------------------------------

typedef __attribute__((ext_vector_type(16))) _Float16     v16h;
typedef __attribute__((ext_vector_type(8)))  float        v8f;
typedef __attribute__((ext_vector_type(4)))  unsigned int v4u;

__device__ inline v8f vzero8() {
  v8f z;
#pragma unroll
  for (int i = 0; i < 8; ++i) z[i] = 0.0f;
  return z;
}

__device__ inline v8f wmma_f16(v16h a, v16h b, v8f c) {
  // D = A(16x32 f16) * B(32x16 f16) + C(16x16 f32)
  return __builtin_amdgcn_wmma_f32_16x16x32_f16(
      /*neg_a=*/false, a, /*neg_b=*/false, b,
      /*c_mod=*/(short)0, c, /*reuse_a=*/false, /*reuse_b=*/false);
}

// --- A fragment (16x32, MxK) from row-major f16 LDS panel -------------------
// Lane l: M = l&15, g = l>>4.  VGPR v holds K = 2v + 16g, pairs.
__device__ inline v16h load_a16(const _Float16* A, int S, int kbase) {
  const int l = threadIdx.x & 31;
  const int M = l & 15, g = l >> 4;
  const unsigned int* p = (const unsigned int*)(A + M * S + kbase);
  union { v16h h; unsigned int u[8]; } r;
#pragma unroll
  for (int v = 0; v < 8; ++v) {
    const int kb = 2 * v + 16 * g;
    r.u[v] = p[kb >> 1];               // ds_load_b32: halves kb, kb+1
  }
  return r.h;
}

// --- B fragment from pre-swizzled f16 weights in d_ws -----------------------
// Fragment = 32 lanes * 16 halves contiguous (1 KB). Two b128 loads per lane.
__device__ inline v16h load_b_ws(const _Float16* frag) {
  const int l = threadIdx.x & 31;
  const v4u* p = (const v4u*)(frag + l * 16);
  union { v16h h; v4u q[2]; } r;
  r.q[0] = p[0];
  r.q[1] = p[1];
  return r.h;
}

// --- B fragment = K^T (32x16, contraction over contiguous dim) from LDS -----
// Lane l: column n = l&15 (key token), K halves contiguous in memory.
__device__ inline v16h load_bT_lds(const _Float16* K, int coloff) {
  const int l = threadIdx.x & 31;
  const int s = l & 15, g = l >> 4;
  const v4u* p = (const v4u*)(K + s * 512 + coloff + 16 * g);
  union { v16h h; v4u q[2]; } r;
  r.q[0] = p[0];
  r.q[1] = p[1];
  return r.h;
}

// --- B fragment = V (16x16 real K, padded to 32) from row-major LDS ---------
__device__ inline v16h load_bV_lds(const _Float16* V, int col) {
  const int l = threadIdx.x & 31;
  const int n = l & 15, g = l >> 4;
  v16h r;
#pragma unroll
  for (int i = 0; i < 16; ++i) {
    _Float16 v = (_Float16)0.0f;
    if (g == 0) v = V[i * 512 + col + n];   // lanes 16-31 = K 16..31 = zero pad
    r[i] = v;
  }
  return r;
}

// --- multi-tile row panel: C[t] += A(16xK) * B_tile(nt0+t) ------------------
template <int NT>
__device__ inline void panel(v8f* c, const _Float16* A, int S,
                             const _Float16* wsb, int nt0, int KT,
                             int kt0, int nkt) {
  for (int k = 0; k < nkt; ++k) {
    v16h a = load_a16(A, S, k * 32);
#pragma unroll
    for (int t = 0; t < NT; ++t) {
      v16h b = load_b_ws(wsb + ((size_t)(nt0 + t) * KT + (kt0 + k)) * 512);
      c[t] = wmma_f16(a, b, c[t]);
    }
  }
}

// --- LayerNorm over 16 rows x 512 cols, 128 threads -------------------------
__device__ inline void layernorm_16x512(const float* xres, _Float16* hdst,
                                        const float* gamma, const float* beta,
                                        float* red, float* stats) {
  const int tid  = threadIdx.x;
  const int row  = tid >> 3;        // 0..15
  const int part = tid & 7;         // 8 threads per row, 64 cols each
  const float* xp = xres + row * 512 + part * 64;
  float s = 0.0f, ss = 0.0f;
#pragma unroll 8
  for (int i = 0; i < 64; ++i) { float v = xp[i]; s += v; ss += v * v; }
  red[(row * 8 + part) * 2 + 0] = s;
  red[(row * 8 + part) * 2 + 1] = ss;
  __syncthreads();
  if (part == 0) {
    float S = 0.0f, SS = 0.0f;
#pragma unroll
    for (int p = 0; p < 8; ++p) {
      S  += red[(row * 8 + p) * 2 + 0];
      SS += red[(row * 8 + p) * 2 + 1];
    }
    const float mu  = S * (1.0f / 512.0f);
    const float var = SS * (1.0f / 512.0f) - mu * mu;
    stats[row * 2 + 0] = mu;
    stats[row * 2 + 1] = rsqrtf(var + 1e-5f);
  }
  __syncthreads();
  const float mu = stats[row * 2 + 0];
  const float rs = stats[row * 2 + 1];
  _Float16* hp = hdst + row * 512 + part * 64;
#pragma unroll 8
  for (int i = 0; i < 64; ++i) {
    const int c = part * 64 + i;
    hp[i] = (_Float16)((xp[i] - mu) * rs * gamma[c] + beta[c]);
  }
}

// ---------------------------------------------------------------------------
// Weight conversion: fp32 -> f16, swizzled into B-fragment layout in d_ws.
// Fragment f (512 halves): lane l, half i -> W[kt*32 + 16*(l>>4) + i][nt*16 + (l&15)]
// Regions (fragment index): [0,512) Wq | [512,1024) Wk | [1024,1536) Wv |
//   [1536,2048) Wo | [2048,4096) W1 (KT=16) | [4096,6144) W2 (KT=64)
// ---------------------------------------------------------------------------
__global__ void __launch_bounds__(256) convert_weights(
    const float* __restrict__ Wq, const float* __restrict__ Wk,
    const float* __restrict__ Wv, const float* __restrict__ Wo,
    const float* __restrict__ W1, const float* __restrict__ W2,
    _Float16* __restrict__ ws) {
  const int gid = blockIdx.x * 256 + threadIdx.x;   // 6144 frags * 32 lanes
  if (gid >= 6144 * 32) return;
  const int frag = gid >> 5, lane = gid & 31;
  const int nl = lane & 15, kb = (lane >> 4) * 16;
  _Float16* dst = ws + ((size_t)frag << 9) + lane * 16;

  if (frag < 2048) {                      // Wq / Wk / Wv / Wo (512x512, KT=16)
    const int m  = frag >> 9;
    const int fl = frag & 511;
    const int nt = fl >> 4, kt = fl & 15;
    const int n  = nt * 16 + nl;
    const int k0 = kt * 32 + kb;
    const float* src = (m == 0) ? Wq : (m == 1) ? Wk : (m == 2) ? Wv : Wo;
#pragma unroll
    for (int i = 0; i < 16; ++i) {
      const int k = k0 + i;
      float v;
      if (m < 3)  // (H, D, 64): element (k=d, n=h*64+e)
        v = src[(size_t)(n >> 6) * 32768 + (size_t)k * 64 + (n & 63)];
      else        // row-major 512x512
        v = src[(size_t)k * 512 + n];
      dst[i] = (_Float16)v;
    }
  } else if (frag < 4096) {               // W1 (512x2048, KT=16)
    const int fl = frag - 2048;
    const int nt = fl >> 4, kt = fl & 15;
    const int n  = nt * 16 + nl;
    const int k0 = kt * 32 + kb;
#pragma unroll
    for (int i = 0; i < 16; ++i)
      dst[i] = (_Float16)W1[(size_t)(k0 + i) * 2048 + n];
  } else {                                // W2 (2048x512, KT=64)
    const int fl = frag - 4096;
    const int nt = fl >> 6, kt = fl & 63;
    const int n  = nt * 16 + nl;
    const int k0 = kt * 32 + kb;
#pragma unroll
    for (int i = 0; i < 16; ++i)
      dst[i] = (_Float16)W2[(size_t)(k0 + i) * 512 + n];
  }
}

// ---------------------------------------------------------------------------
// Main kernel: one workgroup (128 threads = 4 wave32) per batch element.
// ---------------------------------------------------------------------------
__global__ void __launch_bounds__(128) transformer_block(
    const float* __restrict__ x,
    const float* __restrict__ bq, const float* __restrict__ bk,
    const float* __restrict__ bv, const float* __restrict__ bo,
    const float* __restrict__ ln1g, const float* __restrict__ ln1b,
    const float* __restrict__ ln2g, const float* __restrict__ ln2b,
    const float* __restrict__ b1, const float* __restrict__ b2,
    const _Float16* __restrict__ ws, float* __restrict__ out) {
  __shared__ __align__(16) unsigned char smem[103552];
  float*    xres = (float*)smem;                     // 16x512 fp32 residual
  _Float16* hbuf = (_Float16*)(smem + 32768);        // 16x512 f16 (h / ctx / h2)
  _Float16* qbuf = (_Float16*)(smem + 49152);        // 16x512 f16 Q   } ffbuf
  _Float16* kbuf = (_Float16*)(smem + 65536);        // 16x512 f16 K   } 16x1024
  _Float16* vbuf = (_Float16*)(smem + 81920);        // 16x512 f16 V
  float*    red  = (float*)(smem + 98304);           // 16x8x2 LN partials
  float*    stats= (float*)(smem + 99328);           // 16x2  LN mu/rsig
  _Float16* pscr = (_Float16*)(smem + 99456);        // 4 waves x 16x32 probs
  _Float16* ffbuf = qbuf;                            // 16x1024, aliases Q|K

  const int bidx = blockIdx.x;
  const int tid  = threadIdx.x;
  const int w    = tid >> 5;            // wave 0..3
  const int l    = tid & 31;
  const int g    = l >> 4;              // half-wave
  const int N    = l & 15;              // fragment column within a tile

  // ---- phase 0: load x[b] (16x512 fp32) into LDS ---------------------------
  {
    const float4* xg = (const float4*)(x + (size_t)bidx * 8192);
    float4* xs = (float4*)xres;
    for (int i = tid; i < 2048; i += 128) xs[i] = xg[i];
  }
  __syncthreads();

  // ---- phase 1: LN1 -> hbuf (f16) ------------------------------------------
  layernorm_16x512(xres, hbuf, ln1g, ln1b, red, stats);
  __syncthreads();

  // ---- phase 2: Q/K/V = h @ W{q,k,v} + b, f16 into LDS ---------------------
  for (int m = 0; m < 3; ++m) {
    const _Float16* wsb = ws + (size_t)m * 262144;
    _Float16* dst = (m == 0) ? qbuf : (m == 1) ? kbuf : vbuf;
    const float* bias = (m == 0) ? bq : (m == 1) ? bk : bv;
    v8f c[8];
#pragma unroll
    for (int t = 0; t < 8; ++t) c[t] = vzero8();
    panel<8>(c, hbuf, 512, wsb, 8 * w, 16, 0, 16);
#pragma unroll
    for (int t = 0; t < 8; ++t) {
      const int col = (8 * w + t) * 16 + N;
      const float bb = bias[col];
#pragma unroll
      for (int r = 0; r < 8; ++r)
        dst[(r + 8 * g) * 512 + col] = (_Float16)(c[t][r] + bb);
    }
  }
  __syncthreads();

  // ---- phase 3: causal attention, heads 2w and 2w+1; ctx -> hbuf -----------
  {
    _Float16* ps = pscr + w * 512;                   // wave-local 16x32
    for (int i = l; i < 512; i += 32) ps[i] = (_Float16)0.0f;  // zero K-pad
    for (int hh = 2 * w; hh < 2 * w + 2; ++hh) {
      const int coff = hh * 64;
      // scores = Q(16x64) @ K^T(64x16), two K-steps
      v8f s = vzero8();
#pragma unroll
      for (int kt = 0; kt < 2; ++kt) {
        v16h a = load_a16(qbuf, 512, coff + kt * 32);
        v16h b = load_bT_lds(kbuf, coff + kt * 32);
        s = wmma_f16(a, b, s);
      }
      // masked softmax in C-fragment layout (row M = r + 8g, col N = lane)
#pragma unroll
      for (int r = 0; r < 8; ++r) {
        const int M = r + 8 * g;
        float sv = s[r] * 0.125f;                    // 1/sqrt(64)
        sv = (N <= M) ? sv : -1e30f;
        float mx = sv;
#pragma unroll
        for (int d = 8; d; d >>= 1) mx = fmaxf(mx, __shfl_xor(mx, d, 16));
        float p = __expf(sv - mx);
        float sum = p;
#pragma unroll
        for (int d = 8; d; d >>= 1) sum += __shfl_xor(sum, d, 16);
        p *= __frcp_rn(sum);
        ps[M * 32 + N] = (_Float16)p;                // K<16 region only
      }
      // ctx = P(16x16, zero-padded to K=32) @ V(16x64) -> hbuf columns
      v16h a = load_a16(ps, 32, 0);
#pragma unroll
      for (int nt = 0; nt < 4; ++nt) {
        v16h b = load_bV_lds(vbuf, coff + nt * 16);
        v8f c = wmma_f16(a, b, vzero8());
        const int col = coff + nt * 16 + N;
#pragma unroll
        for (int r = 0; r < 8; ++r)
          hbuf[(r + 8 * g) * 512 + col] = (_Float16)c[r];
      }
    }
  }
  __syncthreads();

  // ---- phase 4: x2 = x + ctx @ Wo + bo (fp32, in place in xres) ------------
  {
    v8f c[8];
#pragma unroll
    for (int t = 0; t < 8; ++t) c[t] = vzero8();
    panel<8>(c, hbuf, 512, ws + 786432, 8 * w, 16, 0, 16);
#pragma unroll
    for (int t = 0; t < 8; ++t) {
      const int col = (8 * w + t) * 16 + N;
      const float bb = bo[col];
#pragma unroll
      for (int r = 0; r < 8; ++r) {
        const int idx = (r + 8 * g) * 512 + col;
        xres[idx] = xres[idx] + c[t][r] + bb;
      }
    }
  }
  __syncthreads();

  // ---- phase 5: LN2 -> hbuf (f16) ------------------------------------------
  layernorm_16x512(xres, hbuf, ln2g, ln2b, red, stats);
  __syncthreads();

  // ---- phase 6/7: FF in two 1024-wide halves; FF2 accumulates in regs ------
  v8f acc2[8];
#pragma unroll
  for (int t = 0; t < 8; ++t) acc2[t] = vzero8();

  for (int half = 0; half < 2; ++half) {
    // FF1: 16 tiles per wave -> relu -> ffbuf (16x1024 f16)
    for (int sub = 0; sub < 2; ++sub) {
      const int nt0 = half * 64 + w * 16 + sub * 8;
      v8f c1[8];
#pragma unroll
      for (int t = 0; t < 8; ++t) c1[t] = vzero8();
      panel<8>(c1, hbuf, 512, ws + 1048576, nt0, 16, 0, 16);
#pragma unroll
      for (int t = 0; t < 8; ++t) {
        const int gcol = (nt0 + t) * 16 + N;
        const float bb = b1[gcol];
        const int lcol = gcol - half * 1024;
#pragma unroll
        for (int r = 0; r < 8; ++r)
          ffbuf[(r + 8 * g) * 1024 + lcol] =
              (_Float16)fmaxf(c1[t][r] + bb, 0.0f);
      }
    }
    __syncthreads();
    // FF2 partial: acc2 += relu @ W2[half*1024 : +1024]
    panel<8>(acc2, ffbuf, 1024, ws + 2097152, 8 * w, 64, half * 32, 32);
    __syncthreads();   // ffbuf reused next half
  }

  // ---- phase 8: out = x2 + ff + b2 -----------------------------------------
#pragma unroll
  for (int t = 0; t < 8; ++t) {
    const int col = (8 * w + t) * 16 + N;
    const float bb = b2[col];
#pragma unroll
    for (int r = 0; r < 8; ++r) {
      const int M = r + 8 * g;
      out[((size_t)bidx * 16 + M) * 512 + col] =
          xres[M * 512 + col] + acc2[t][r] + bb;
    }
  }
}

// ---------------------------------------------------------------------------
extern "C" void kernel_launch(void* const* d_in, const int* in_sizes, int n_in,
                              void* d_out, int out_size, void* d_ws, size_t ws_size,
                              hipStream_t stream) {
  const float* x    = (const float*)d_in[0];
  const float* Wq   = (const float*)d_in[1];
  const float* bq   = (const float*)d_in[2];
  const float* Wk   = (const float*)d_in[3];
  const float* bk   = (const float*)d_in[4];
  const float* Wv   = (const float*)d_in[5];
  const float* bv   = (const float*)d_in[6];
  const float* Wo   = (const float*)d_in[7];
  const float* bo   = (const float*)d_in[8];
  const float* ln1g = (const float*)d_in[9];
  const float* ln1b = (const float*)d_in[10];
  const float* ln2g = (const float*)d_in[11];
  const float* ln2b = (const float*)d_in[12];
  const float* W1   = (const float*)d_in[13];
  const float* b1   = (const float*)d_in[14];
  const float* W2   = (const float*)d_in[15];
  const float* b2   = (const float*)d_in[16];
  _Float16* ws = (_Float16*)d_ws;
  float* out = (float*)d_out;

  // 6144 fragments * 32 lanes = 196608 work items
  convert_weights<<<768, 256, 0, stream>>>(Wq, Wk, Wv, Wo, W1, W2, ws);
  transformer_block<<<8192, 128, 0, stream>>>(
      x, bq, bk, bv, bo, ln1g, ln1b, ln2g, ln2b, b1, b2, ws, out);
}
